// UNetEVTransformerBlock_65438121722510
// MI455X (gfx1250) — matrix operand order repaired and checked by reading, not compiled, not run on hardware
//
#include <hip/hip_runtime.h>
#include <hip/hip_bf16.h>

// MI455X / gfx1250, wave32. All GEMMs via v_wmma_f32_16x16x32_f16.
//
// Pipeline (B=4, C=128, H=W=128, HW=16384):
//  stage_a : per 128-token tile: LN + (w_proj GEMM + GELU) -> masked per-channel
//            partial sums (deterministic slots, no float atomics)
//  stage_b : scale = ratio @ exp(Wc).T, global_x = partials/denom, sigmoid(alpha_eff)
//  stage_c : recompute local tile, ChannelPool branch, token MLP 256->128->64->2
//            (WMMA), scores + weighting
//  stage_d : per-batch radix-select top-k (k=6553) hi & lo -> new_mask

typedef __attribute__((ext_vector_type(16))) _Float16 h16;
typedef __attribute__((ext_vector_type(8)))  _Float16 h8;
typedef __attribute__((ext_vector_type(8)))  float    f8;

#define PITCH 144   // f16 pitch, multiple of 16 (32B) for aligned fragments
#define HW_N  16384
#define C_N   128

__device__ __forceinline__ float gelu_f(float x) {
    return 0.5f * x * (1.0f + erff(x * 0.70710678118654752f));
}

// A-fragment: row-major f16 W[pitch]; lane m = lane&15, K = 16*(lane>=16)+j
__device__ __forceinline__ h16 load_a_frag(const _Float16* __restrict__ W,
                                           int m16, int k, int lane) {
    int row = m16 + (lane & 15);
    int k8  = k + ((lane & 16) ? 16 : 0);
    const _Float16* p = W + row * PITCH + k8;
    return *(const h16*)p;
}

// B-fragment: token-major f16 Bt[pitch]; lane n = lane&15, K = 16*(lane>=16)+j
__device__ __forceinline__ h16 load_b_frag(const _Float16* __restrict__ Bt,
                                           int k, int n16, int lane) {
    int row = n16 + (lane & 15);
    int kk  = k + ((lane & 16) ? 16 : 0);
    return *(const h16*)(Bt + row * PITCH + kk);
}

// C tile -> GELU -> packed f16 token-major store (one 16B ds_store per lane/tile)
__device__ __forceinline__ void store_c_gelu(const f8& acc, _Float16* __restrict__ Ot,
                                             int opitch, int m16, int n16, int lane) {
    int row = n16 + (lane & 15);
    int cb  = m16 + ((lane & 16) ? 8 : 0);
    h8 p;
#pragma unroll
    for (int r = 0; r < 8; ++r) p[r] = (_Float16)gelu_f(acc[r]);
    *(h8*)(Ot + row * opitch + cb) = p;
}

template <int NT, int KC>
__device__ __forceinline__ void gemm_acc(f8* acc, const _Float16* __restrict__ A,
                                         const _Float16* __restrict__ Bt,
                                         int m16, int nbase, int lane) {
#pragma unroll
    for (int k = 0; k < KC; k += 32) {
        h16 a = load_a_frag(A, m16, k, lane);
#pragma unroll
        for (int n = 0; n < NT; ++n) {
            h16 b = load_b_frag(Bt, k, nbase + n * 16, lane);
            acc[n] = __builtin_amdgcn_wmma_f32_16x16x32_f16(
                false, a, false, b, (short)0, acc[n], false, false);
        }
    }
}

// LayerNorm over channels + local_x = GELU(w_proj @ xn) for a 128-token tile.
// Output LXT: token-major f16 [128][PITCH].
// MUST be force-inlined so LDS pointers keep addrspace(3) (no generic-pointer
// null-check cndmask chains around the WMMA fragment ds_loads).
__device__ __forceinline__ void tile_local(
        const float* __restrict__ x,
        const float* __restrict__ lnw_g, const float* __restrict__ lnb_g,
        const float* __restrict__ wproj,
        _Float16* __restrict__ WH, _Float16* __restrict__ XNT,
        _Float16* __restrict__ LXT,
        float* __restrict__ MU, float* __restrict__ INVS,
        float* __restrict__ LNW, float* __restrict__ LNB,
        float* __restrict__ SRED,
        int b, int tok0) {
    const int tid = threadIdx.x;
    // stage weights (f32 -> f16 LDS, row-major) + LN params
    for (int idx = tid; idx < C_N * C_N; idx += 256) {
        int o = idx >> 7, i = idx & 127;
        WH[o * PITCH + i] = (_Float16)wproj[idx];
    }
    if (tid < C_N) { LNW[tid] = lnw_g[tid]; LNB[tid] = lnb_g[tid]; }
    // LN stats: coalesced global reads, each half-warp-set covers half the channels
    {
        int t = tid & 127, half = tid >> 7;
        const float* xb = x + (long)b * C_N * HW_N + tok0 + t;
        float s = 0.f, q = 0.f;
        int c0 = half * 64;
        for (int c = c0; c < c0 + 64; ++c) {
            float v = xb[(long)c * HW_N];
            s += v; q += v * v;
        }
        SRED[half * 128 + t] = s;
        SRED[256 + half * 128 + t] = q;
    }
    __syncthreads();
    if (tid < C_N) {
        float s = SRED[tid] + SRED[128 + tid];
        float q = SRED[256 + tid] + SRED[384 + tid];
        float mu = s * (1.f / 128.f);
        float var = q * (1.f / 128.f) - mu * mu;
        MU[tid] = mu;
        INVS[tid] = rsqrtf(var + 1e-5f);
    }
    __syncthreads();
    // normalize -> token-major f16 (global reads hit cache from pass 1)
    for (int idx = tid; idx < C_N * C_N; idx += 256) {
        int c = idx >> 7, t = idx & 127;
        float v = x[((long)b * C_N + c) * HW_N + tok0 + t];
        XNT[t * PITCH + c] = (_Float16)((v - MU[t]) * INVS[t] * LNW[c] + LNB[c]);
    }
    __syncthreads();
    // 128x128x128 GEMM: 8 waves x (16 rows x 128 tokens), 4 k-steps of 32
    const int wave = tid >> 5, lane = tid & 31;
    f8 acc[8] = {};
    gemm_acc<8, 128>(acc, WH, XNT, wave * 16, 0, lane);
#pragma unroll
    for (int n = 0; n < 8; ++n) store_c_gelu(acc[n], LXT, PITCH, wave * 16, n * 16, lane);
    __syncthreads();
}

__global__ __launch_bounds__(256) void stage_a(const float* __restrict__ x,
                                               const float* __restrict__ maskp,
                                               const float* __restrict__ lnw,
                                               const float* __restrict__ lnb,
                                               const float* __restrict__ wproj,
                                               float* __restrict__ part) {
    __shared__ __attribute__((aligned(32))) _Float16 WH[C_N * PITCH];
    __shared__ __attribute__((aligned(32))) _Float16 XNT[C_N * PITCH];
    __shared__ __attribute__((aligned(32))) _Float16 LXT[C_N * PITCH];
    __shared__ float MU[128], INVS[128], LNW[128], LNB[128], MK[128], SRED[512];

    const int tid = threadIdx.x;
    const int b = blockIdx.y, blk = blockIdx.x, tok0 = blk * 128;
    if (tid < 128) MK[tid] = maskp[b * HW_N + tok0 + tid];
    tile_local(x, lnw, lnb, wproj, WH, XNT, LXT, MU, INVS, LNW, LNB, SRED, b, tok0);
    // masked per-channel partial sums -> deterministic slot per (b, blk)
    if (tid < 128) {
        int c = tid;
        float s = 0.f;
        for (int t = 0; t < 128; ++t) s += (float)LXT[t * PITCH + c] * MK[t];
        part[((long)b * 128 + blk) * 128 + c] = s;
    }
}

__global__ __launch_bounds__(256) void stage_b(const float* __restrict__ ratio,
                                               const float* __restrict__ wctrl,
                                               const float* __restrict__ maskp,
                                               const float* __restrict__ alpha,
                                               const float* __restrict__ part,
                                               float* __restrict__ gx,
                                               float* __restrict__ scaleo,
                                               float* __restrict__ sigo) {
    __shared__ float RED[256], DEN[4];
    const int tid = threadIdx.x;
    {   // denom[b] = sum(mask[b])
        int b = tid >> 6, l = tid & 63;
        float s = 0.f;
        for (int i = l; i < HW_N; i += 64) s += maskp[b * HW_N + i];
        RED[tid] = s;
    }
    __syncthreads();
    if ((tid & 63) == 0) {
        float s = 0.f;
        for (int j = 0; j < 64; ++j) s += RED[(tid & 192) + j];
        DEN[tid >> 6] = s;
    }
    __syncthreads();
    // PositiveLinear: scale[b][j] = sum_r ratio[b][r]*exp(Wc[j][r])
    for (int idx = tid; idx < 1024; idx += 256) {
        int b = idx >> 8, j = idx & 255;
        float s = 0.f;
        for (int r = 0; r < 6; ++r) s += ratio[b * 6 + r] * expf(wctrl[j * 6 + r]);
        scaleo[idx] = s;
    }
    // global_x[b][c]
    for (int idx = tid; idx < 512; idx += 256) {
        int b = idx >> 7, c = idx & 127;
        float s = 0.f;
        for (int blk = 0; blk < 128; ++blk) s += part[((long)b * 128 + blk) * 128 + c];
        float g = s / DEN[b];
        if (g == INFINITY) g = 0.f;
        gx[idx] = g;
    }
    if (tid == 0) {
        float am = alpha[0];
        for (int i = 0; i < 24; ++i) am = fmaxf(am, ratio[i]);
        sigo[0] = 1.f / (1.f + expf(-am));
    }
}

__global__ __launch_bounds__(256) void stage_c(const float* __restrict__ x,
                                               const float* __restrict__ lnw,
                                               const float* __restrict__ lnb,
                                               const float* __restrict__ wproj,
                                               const float* __restrict__ w1,
                                               const float* __restrict__ w2,
                                               const float* __restrict__ w3,
                                               const float* __restrict__ w4,
                                               const float* __restrict__ wcc,
                                               const float* __restrict__ gxg,
                                               const float* __restrict__ scg,
                                               const float* __restrict__ sigg,
                                               float* __restrict__ score,
                                               float* __restrict__ wout) {
    // P0: wproj -> vt-half -> h2t | P1: xnt -> h1t | P2: lxt -> w2h | P3: w1-half
    __shared__ __attribute__((aligned(32))) _Float16 P0[C_N * PITCH];
    __shared__ __attribute__((aligned(32))) _Float16 P1[C_N * PITCH];
    __shared__ __attribute__((aligned(32))) _Float16 P2[C_N * PITCH];
    __shared__ __attribute__((aligned(32))) _Float16 P3[C_N * PITCH];
    __shared__ float MU[128], INVS[128], LNW[128], LNB[128];
    __shared__ float GX[128], SC[256], X1A[128], X1B[128], SRED[512];
    __shared__ _Float16 GXS[128];

    const int tid = threadIdx.x, wave = tid >> 5, lane = tid & 31;
    const int b = blockIdx.y, tok0 = blockIdx.x * 128;

    // preload w1 K-half 1 (cols 0..127) while LN/local GEMM proceeds
    for (int idx = tid; idx < 16384; idx += 256) {
        int j = idx >> 7, k = idx & 127;
        P3[j * PITCH + k] = (_Float16)w1[j * 256 + k];
    }
    if (tid < 128) GX[tid] = gxg[b * 128 + tid];
    SC[tid] = scg[b * 256 + tid];

    tile_local(x, lnw, lnb, wproj, P0, P1, P2, MU, INVS, LNW, LNB, SRED, b, tok0);

    if (tid < 128) GXS[tid] = (_Float16)(GX[tid] * SC[128 + tid]);
    // ChannelPool(max,mean) over t = local*global, then 2x2 conv + GELU
    if (tid < 128) {
        int t = tid;
        float mx = -INFINITY, sm = 0.f;
        for (int c = 0; c < 128; ++c) {
            float v = (float)P2[t * PITCH + c] * GX[c];
            mx = fmaxf(mx, v);
            sm += v;
        }
        float mn = sm * (1.f / 128.f);
        X1A[t] = gelu_f(wcc[0] * mx + wcc[1] * mn);
        X1B[t] = gelu_f(wcc[2] * mx + wcc[3] * mn);
    }
    // vt K-half 1: local_x * scale[0:128]  (P0: wproj is dead)
    for (int idx = tid; idx < 16384; idx += 256) {
        int t = idx & 127, c = idx >> 7;
        P0[t * PITCH + c] = (_Float16)((float)P2[t * PITCH + c] * SC[c]);
    }
    // prefetch upcoming weight tiles into cache
    if (tid < 128) { __builtin_prefetch(&w1[tid * 256 + 128]); __builtin_prefetch(&w1[tid * 256 + 192]); }
    if (tid < 64)  { __builtin_prefetch(&w2[tid * 128]); __builtin_prefetch(&w2[tid * 128 + 64]); }
    __syncthreads();

    // GEMM1 (256->128), K split across two LDS fills
    f8 acc[8] = {};
    gemm_acc<8, 128>(acc, P3, P0, wave * 16, 0, lane);
    __syncthreads();
    // vt K-half 2: broadcast global_x*scale[128:256]; w1 K-half 2; w2
    for (int idx = tid; idx < 16384; idx += 256) {
        int t = idx & 127, c = idx >> 7;
        P0[t * PITCH + c] = GXS[c];
    }
    for (int idx = tid; idx < 16384; idx += 256) {
        int j = idx >> 7, k = idx & 127;
        P3[j * PITCH + k] = (_Float16)w1[j * 256 + 128 + k];
    }
    for (int idx = tid; idx < 8192; idx += 256) {
        int j = idx >> 7, k = idx & 127;
        P2[j * PITCH + k] = (_Float16)w2[idx];
    }
    __syncthreads();
    gemm_acc<8, 128>(acc, P3, P0, wave * 16, 0, lane);
#pragma unroll
    for (int n = 0; n < 8; ++n) store_c_gelu(acc[n], P1, PITCH, wave * 16, n * 16, lane);
    __syncthreads();

    // GEMM2 (128->64): M=64 -> 4 row-blocks, N split across wave pairs
    f8 acc2[4] = {};
    gemm_acc<4, 128>(acc2, P2, P1, (wave & 3) * 16, (wave >> 2) * 64, lane);
#pragma unroll
    for (int n = 0; n < 4; ++n)
        store_c_gelu(acc2[n], P0, 72, (wave & 3) * 16, (wave >> 2) * 64 + n * 16, lane);
    __syncthreads();

    // tail: 64->2, concat with branch-1, 4->1, scores + weighting
    if (tid < 128) {
        int t = tid;
        float h3a = 0.f, h3b = 0.f;
        for (int i = 0; i < 64; ++i) {
            float v = (float)P0[t * 72 + i];
            h3a += w3[i] * v;
            h3b += w3[64 + i] * v;
        }
        h3a = gelu_f(h3a);
        h3b = gelu_f(h3b);
        float pre = w4[0] * X1A[t] + w4[1] * X1B[t] + w4[2] * h3a + w4[3] * h3b;
        float sv = gelu_f(pre);
        long o = (long)b * HW_N + tok0 + t;
        score[o] = sv;
        wout[o] = sigg[0] * gelu_f(sv);
    }
}

__device__ __forceinline__ unsigned key_of(float v) {
    unsigned u = __float_as_uint(v);
    return (u & 0x80000000u) ? ~u : (u | 0x80000000u);
}

__global__ __launch_bounds__(256) void stage_d(const float* __restrict__ score,
                                               float* __restrict__ maskout, int k) {
    __shared__ unsigned HIST[256];
    __shared__ unsigned SB[2];
    const int b = blockIdx.x, tid = threadIdx.x;
    const float* s = score + (long)b * HW_N;

    unsigned thi, tlo;
    {   // k-th largest (8-bit MSD radix select, 4 passes)
        unsigned prefix = 0, pmask = 0; int kk = k;
        for (int shift = 24; shift >= 0; shift -= 8) {
            HIST[tid] = 0; __syncthreads();
            for (int i = tid; i < HW_N; i += 256) {
                unsigned u = key_of(s[i]);
                if ((u & pmask) == prefix) atomicAdd(&HIST[(u >> shift) & 255], 1u);
            }
            __syncthreads();
            if (tid == 0) {
                int cum = 0, binsel = 0;
                for (int bin = 255; bin >= 0; --bin) {
                    int c = (int)HIST[bin];
                    if (cum + c >= kk) { binsel = bin; break; }
                    cum += c;
                }
                SB[0] = (unsigned)binsel; SB[1] = (unsigned)(kk - cum);
            }
            __syncthreads();
            prefix |= SB[0] << shift; pmask |= 0xFFu << shift; kk = (int)SB[1];
            __syncthreads();
        }
        thi = prefix;
    }
    {   // k-th smallest
        unsigned prefix = 0, pmask = 0; int kk = k;
        for (int shift = 24; shift >= 0; shift -= 8) {
            HIST[tid] = 0; __syncthreads();
            for (int i = tid; i < HW_N; i += 256) {
                unsigned u = key_of(s[i]);
                if ((u & pmask) == prefix) atomicAdd(&HIST[(u >> shift) & 255], 1u);
            }
            __syncthreads();
            if (tid == 0) {
                int cum = 0, binsel = 0;
                for (int bin = 0; bin < 256; ++bin) {
                    int c = (int)HIST[bin];
                    if (cum + c >= kk) { binsel = bin; break; }
                    cum += c;
                }
                SB[0] = (unsigned)binsel; SB[1] = (unsigned)(kk - cum);
            }
            __syncthreads();
            prefix |= SB[0] << shift; pmask |= 0xFFu << shift; kk = (int)SB[1];
            __syncthreads();
        }
        tlo = prefix;
    }
    for (int i = tid; i < HW_N; i += 256) {
        unsigned u = key_of(s[i]);
        maskout[(long)b * HW_N + i] = (u >= thi || u <= tlo) ? 1.f : 0.f;
    }
}

extern "C" void kernel_launch(void* const* d_in, const int* in_sizes, int n_in,
                              void* d_out, int out_size, void* d_ws, size_t ws_size,
                              hipStream_t stream) {
    const float* x     = (const float*)d_in[0];
    const float* maskp = (const float*)d_in[1];
    const float* ratio = (const float*)d_in[2];
    const float* lnw   = (const float*)d_in[3];
    const float* lnb   = (const float*)d_in[4];
    const float* wproj = (const float*)d_in[5];
    const float* wctrl = (const float*)d_in[6];
    const float* wcc   = (const float*)d_in[7];
    const float* w1    = (const float*)d_in[8];
    const float* w2    = (const float*)d_in[9];
    const float* w3    = (const float*)d_in[10];
    const float* w4    = (const float*)d_in[11];
    const float* alpha = (const float*)d_in[12];

    float* out = (float*)d_out;          // [0,65536): new_mask, [65536,131072): weighting
    float* ws  = (float*)d_ws;           // ~532 KB used
    float* part  = ws;                   // 65536
    float* gx    = ws + 65536;           // 512
    float* scale = ws + 66048;           // 1024
    float* sig   = ws + 67072;           // 1
    float* score = ws + 67584;           // 65536

    dim3 blk(256);
    dim3 grid(128, 4);
    stage_a<<<grid, blk, 0, stream>>>(x, maskp, lnw, lnb, wproj, part);
    stage_b<<<1, blk, 0, stream>>>(ratio, wctrl, maskp, alpha, part, gx, scale, sig);
    stage_c<<<grid, blk, 0, stream>>>(x, lnw, lnb, wproj, w1, w2, w3, w4, wcc,
                                      gx, scale, sig, score, out + 65536);
    stage_d<<<4, blk, 0, stream>>>(score, out, 6553);   // k = int(0.4*16384)
}
